// CausalSelfAttention_51642686767796
// MI455X (gfx1250) — matrix operand order repaired and checked by reading, not compiled, not run on hardware
//
#include <hip/hip_runtime.h>

#define DIM   2048
#define NH    16
#define NKV   4
#define HD    128

typedef __attribute__((ext_vector_type(16))) __bf16 v16bf;
typedef __attribute__((ext_vector_type(8)))  __bf16 v8bf;
typedef __attribute__((ext_vector_type(8)))  float  v8f;
typedef unsigned int v4u __attribute__((ext_vector_type(4)));
typedef int          v4i __attribute__((ext_vector_type(4)));
typedef int          v8i __attribute__((ext_vector_type(8)));

#ifndef __has_builtin
#define __has_builtin(x) 0
#endif

// TDM available only in the device pass; arity differs by toolchain.
#if defined(__AMDGCN__) && __has_builtin(__builtin_amdgcn_tensor_load_to_lds) && \
    __has_builtin(__builtin_amdgcn_s_wait_tensorcnt)
#define HAVE_TDM 1
#else
#define HAVE_TDM 0
#endif
#if __has_include(<hip/amd_detail/amd_gfx1250_TDM.h>)
#define TDM_6ARG 1   // therock-10.0 headers => 6-arg builtin
#else
#define TDM_6ARG 0   // ROCm 7.2 => 5-arg builtin
#endif

// ---------------------------------------------------------------------------
// WMMA helper: D = A(16x32 bf16) * B(32x16 bf16) + C(16x16 f32)
// ---------------------------------------------------------------------------
__device__ __forceinline__ v8f wmma_bf16(v16bf a, v16bf b, v8f c) {
  return __builtin_amdgcn_wmma_f32_16x16x32_bf16(
      /*neg_a=*/false, a, /*neg_b=*/false, b,
      /*c_mod=*/(short)0, c, /*reuse_a=*/false, /*reuse_b=*/false);
}

// A fragment (16x32, row-major source, contract along row):
//   lane L: m = L&15, holds K = koff..koff+15, koff=(L>>4)*16
__device__ __forceinline__ v16bf load_a_frag(const __bf16* base, int ld,
                                             int mtile, int kbase, int lane) {
  int m    = lane & 15;
  int koff = (lane >> 4) * 16;
  const __bf16* p = base + (size_t)(mtile + m) * ld + kbase + koff;
  v8bf lo = *(const v8bf*)p;
  v8bf hi = *(const v8bf*)(p + 8);
  v16bf r;
#pragma unroll
  for (int i = 0; i < 8; ++i) { r[i] = lo[i]; r[i + 8] = hi[i]; }
  return r;
}

// B fragment (32x16): source rows are the N dimension, contiguous along K:
//   lane L: n = L&15, holds K = (L>>4)*16 .. +15  (one 32B load)
__device__ __forceinline__ v16bf load_b_frag(const __bf16* base, int ld,
                                             int ntile, int kbase, int lane) {
  int n    = lane & 15;
  int koff = (lane >> 4) * 16;
  const __bf16* p = base + (size_t)(ntile + n) * ld + kbase + koff;
  return *(const v16bf*)p;
}

// ---------------------------------------------------------------------------
// TDM: 2-D tile load (tile_dim0=32 elems wide, tile_dim1=64 rows) of bf16
// tensor [tensor_h rows x tensor_w elems], row stride ld elems, into LDS.
// D# packing per CDNA5 ISA.
// ---------------------------------------------------------------------------
#if HAVE_TDM
__device__ __forceinline__ void tdm_load_tile_2d(const __bf16* gaddr,
                                                 unsigned lds_byte_addr,
                                                 int tensor_w, int tensor_h,
                                                 int ld_elems) {
  unsigned long long ga = (unsigned long long)(__SIZE_TYPE__)gaddr;
  v4u g0;
  g0[0] = 1u;                                        // count=1, user mode
  g0[1] = lds_byte_addr;                             // lds_addr
  g0[2] = (unsigned)(ga & 0xffffffffu);              // global_addr[31:0]
  g0[3] = (unsigned)((ga >> 32) & 0x1ffffffu)        // global_addr[56:32]
          | (2u << 30);                              // type = 2 ("image")
  v8i g1;
  g1[0] = (int)(1u << 16);                           // data_size=1 (2 bytes)
  g1[1] = (int)(((unsigned)tensor_w & 0xffffu) << 16);           // dim0[15:0]
  g1[2] = (int)((((unsigned)tensor_w >> 16) & 0xffffu)           // dim0[31:16]
                | (((unsigned)tensor_h & 0xffffu) << 16));       // dim1[15:0]
  g1[3] = (int)((((unsigned)tensor_h >> 16) & 0xffffu)           // dim1[31:16]
                | (32u << 16));                                  // tile_dim0=32
  g1[4] = 64;                                        // tile_dim1=64, tile_dim2=0
  g1[5] = ld_elems;                                  // dim0_stride[31:0]
  g1[6] = 0;                                         // dim0_stride[47:32], dim1_stride lo
  g1[7] = 0;
  v4i z2 = {0, 0, 0, 0};
#if TDM_6ARG
  v8i z3 = {0, 0, 0, 0, 0, 0, 0, 0};
  __builtin_amdgcn_tensor_load_to_lds(g0, g1, z2, z2, z3, 0);
#else
  __builtin_amdgcn_tensor_load_to_lds(g0, g1, z2, z2, 0);
#endif
}
#endif

// ---------------------------------------------------------------------------
// fp32 -> bf16 conversion
// ---------------------------------------------------------------------------
__global__ void f32_to_bf16_kernel(const float* __restrict__ in,
                                   __bf16* __restrict__ out, int n) {
  int i = blockIdx.x * blockDim.x + threadIdx.x;
  if (i < n) out[i] = (__bf16)in[i];
}

// ---------------------------------------------------------------------------
// C[M,N](f32) = A[M,K](bf16) * B[N,K](bf16)^T
// Block: 4 waves, 64x64 tile; wave: 32x32. TDM double-buffers the 64x32 A/B
// tiles into LDS (one issuing wave, TENSORcnt pipelining); WMMA reads LDS.
// ---------------------------------------------------------------------------
__global__ __launch_bounds__(128)
void gemm_bf16_kernel(const __bf16* __restrict__ A, const __bf16* __restrict__ B,
                      float* __restrict__ C, int M, int N, int K) {
  int lane  = threadIdx.x & 31;
  int wid   = threadIdx.x >> 5;
  int nblock = blockIdx.x * 64;
  int mblock = blockIdx.y * 64;
  int wm = (wid >> 1) * 32;   // wave tile offsets inside the 64x64 block tile
  int wn = (wid & 1) * 32;

  v8f acc[2][2] = {};

#if HAVE_TDM
  __shared__ __bf16 sA[2][64 * 32];
  __shared__ __bf16 sB[2][64 * 32];
  const int nsteps = K / 32;

  unsigned ldsA0 = (unsigned)(__SIZE_TYPE__)&sA[0][0];
  unsigned ldsA1 = (unsigned)(__SIZE_TYPE__)&sA[1][0];
  unsigned ldsB0 = (unsigned)(__SIZE_TYPE__)&sB[0][0];
  unsigned ldsB1 = (unsigned)(__SIZE_TYPE__)&sB[1][0];

  if (wid == 0) {  // prologue: issue tile pair 0
    tdm_load_tile_2d(A + (size_t)mblock * K, ldsA0, K, M, K);
    tdm_load_tile_2d(B + (size_t)nblock * K, ldsB0, K, N, K);
  }

  for (int s = 0; s < nsteps; ++s) {
    int buf = s & 1;
    if (wid == 0) {
      if (s + 1 < nsteps) {  // issue next pair into the other buffer
        const __bf16* An = A + (size_t)mblock * K + (size_t)(s + 1) * 32;
        const __bf16* Bn = B + (size_t)nblock * K + (size_t)(s + 1) * 32;
        tdm_load_tile_2d(An, buf ? ldsA0 : ldsA1, K, M, K);
        tdm_load_tile_2d(Bn, buf ? ldsB0 : ldsB1, K, N, K);
        __builtin_amdgcn_s_wait_tensorcnt(2);  // oldest pair has landed
      } else {
        __builtin_amdgcn_s_wait_tensorcnt(0);
      }
    }
    __syncthreads();  // tiles in LDS visible to all waves

    const __bf16* tA = sA[buf];
    const __bf16* tB = sB[buf];
    v16bf a0 = load_a_frag(tA, 32, wm,      0, lane);
    v16bf a1 = load_a_frag(tA, 32, wm + 16, 0, lane);
    v16bf b0 = load_b_frag(tB, 32, wn,      0, lane);
    v16bf b1 = load_b_frag(tB, 32, wn + 16, 0, lane);
    acc[0][0] = wmma_bf16(a0, b0, acc[0][0]);
    acc[0][1] = wmma_bf16(a0, b1, acc[0][1]);
    acc[1][0] = wmma_bf16(a1, b0, acc[1][0]);
    acc[1][1] = wmma_bf16(a1, b1, acc[1][1]);

    __syncthreads();  // all reads done before next TDM overwrite
  }
#else
  // Fallback: direct global-memory fragments (round-1 path)
  int mtile = mblock + wm, ntile = nblock + wn;
  for (int k = 0; k < K; k += 32) {
    if (k + 64 < K) {
      __builtin_prefetch(A + (size_t)(mtile + (lane & 15)) * K + k + 64, 0, 1);
      __builtin_prefetch(B + (size_t)(ntile + (lane & 15)) * K + k + 64, 0, 1);
    }
    v16bf a0 = load_a_frag(A, K, mtile,      k, lane);
    v16bf a1 = load_a_frag(A, K, mtile + 16, k, lane);
    v16bf b0 = load_b_frag(B, K, ntile,      k, lane);
    v16bf b1 = load_b_frag(B, K, ntile + 16, k, lane);
    acc[0][0] = wmma_bf16(a0, b0, acc[0][0]);
    acc[0][1] = wmma_bf16(a0, b1, acc[0][1]);
    acc[1][0] = wmma_bf16(a1, b0, acc[1][0]);
    acc[1][1] = wmma_bf16(a1, b1, acc[1][1]);
  }
#endif

  int rowoff = (lane < 16) ? 0 : 8;
  int col    = lane & 15;
#pragma unroll
  for (int i = 0; i < 2; ++i)
#pragma unroll
    for (int j = 0; j < 2; ++j)
#pragma unroll
      for (int r = 0; r < 8; ++r) {
        int row = mblock + wm + i * 16 + rowoff + r;
        C[(size_t)row * N + nblock + wn + j * 16 + col] = acc[i][j][r];
      }
}

// ---------------------------------------------------------------------------
// Fused per-head RMSNorm + RoPE + scale, fp32 [T, nh*128] -> bf16 [nh][T][128]
// One wave handles one (t, head); lane owns rope pairs p=2*lane, 2*lane+1.
// ---------------------------------------------------------------------------
__global__ __launch_bounds__(256)
void rope_rms_pack_kernel(const float* __restrict__ in,
                          const float* __restrict__ gain,
                          __bf16* __restrict__ out,
                          int T, int nh, float scale) {
  int wid  = (blockIdx.x * blockDim.x + threadIdx.x) >> 5;
  int lane = threadIdx.x & 31;
  int t = wid / nh;
  int h = wid % nh;
  if (t >= T) return;

  const float* row = in + (size_t)t * (nh * HD) + (size_t)h * HD;
  int p0 = lane * 2;
  float x1a = row[p0],      x1b = row[p0 + 1];
  float x2a = row[64 + p0], x2b = row[64 + p0 + 1];

  float ss = x1a * x1a + x1b * x1b + x2a * x2a + x2b * x2b;
#pragma unroll
  for (int off = 16; off >= 1; off >>= 1) ss += __shfl_xor(ss, off, 32);
  float inv = rsqrtf(ss * (1.0f / 128.0f) + 1.19209290e-07f);
  float g = (gain ? gain[h] : 1.0f) * scale;

  const float LN_BASE = 9.210340371976184f;  // ln(10000)
  float fa = __expf(-(float)(2 * p0)       * (1.0f / 128.0f) * LN_BASE);
  float fb = __expf(-(float)(2 * (p0 + 1)) * (1.0f / 128.0f) * LN_BASE);
  float aa = (float)t * fa, ab = (float)t * fb;
  float ca = __cosf(aa), sa = __sinf(aa);
  float cb = __cosf(ab), sb = __sinf(ab);

  float n1a = x1a * inv, n2a = x2a * inv;
  float n1b = x1b * inv, n2b = x2b * inv;
  float y1a = n1a * ca + n2a * sa, y2a = -n1a * sa + n2a * ca;
  float y1b = n1b * cb + n2b * sb, y2b = -n1b * sb + n2b * cb;

  __bf16* orow = out + ((size_t)h * T + t) * HD;
  orow[p0]          = (__bf16)(y1a * g);
  orow[p0 + 1]      = (__bf16)(y1b * g);
  orow[64 + p0]     = (__bf16)(y2a * g);
  orow[64 + p0 + 1] = (__bf16)(y2b * g);
}

// ---------------------------------------------------------------------------
// vf fp32 [T, NKV*128] -> Vt bf16 [NKV][128][T]  (transpose for P*V B-frags)
// ---------------------------------------------------------------------------
__global__ void v_transpose_kernel(const float* __restrict__ vf,
                                   __bf16* __restrict__ vt, int T) {
  int idx = blockIdx.x * blockDim.x + threadIdx.x;
  if (idx >= T * NKV * HD) return;
  int t = idx >> 9;           // /512
  int c = idx & 511;
  int h = c >> 7;
  int d = c & 127;
  vt[((size_t)h * HD + d) * T + t] = (__bf16)vf[idx];
}

// ---------------------------------------------------------------------------
// Flash attention: one wave per (16-query tile, head). bf16 WMMA, f32 accum,
// online softmax, P transits LDS to convert D-layout -> A-layout.
// Writes aout bf16 [T, 2048]  (t-major, head*128+d columns).
// ---------------------------------------------------------------------------
__global__ __launch_bounds__(32)
void attention_kernel(const __bf16* __restrict__ Qb, const __bf16* __restrict__ Kb,
                      const __bf16* __restrict__ Vt, __bf16* __restrict__ aout,
                      int T) {
  __shared__ __bf16 P[16 * 32];

  int lane  = threadIdx.x;
  int qtile = blockIdx.x * 16;
  int h     = blockIdx.y;
  int hk    = h >> 2;  // rep = NH/NKV = 4

  const __bf16* Qh = Qb + (size_t)h  * T * HD;
  const __bf16* Kh = Kb + (size_t)hk * T * HD;
  const __bf16* Vh = Vt + (size_t)hk * HD * T;

  v16bf qfrag[4];
#pragma unroll
  for (int c = 0; c < 4; ++c) qfrag[c] = load_a_frag(Qh, HD, qtile, c * 32, lane);

  v8f oacc[8] = {};
  float mrow[8], lrow[8];
#pragma unroll
  for (int r = 0; r < 8; ++r) { mrow[r] = -3.0e38f; lrow[r] = 0.0f; }

  int rowoff = (lane < 16) ? 0 : 8;
  int ncol   = lane & 15;
  int kend   = qtile + 16;

  for (int kt = 0; kt < kend; kt += 32) {
    // ---- scores S[16 x 32] = Q * K^T (contract over d=128) ----
    v8f s0 = {}, s1 = {};
#pragma unroll
    for (int c = 0; c < 4; ++c) {
      v16bf b0 = load_b_frag(Kh, HD, kt,      c * 32, lane);
      v16bf b1 = load_b_frag(Kh, HD, kt + 16, c * 32, lane);
      s0 = wmma_bf16(qfrag[c], b0, s0);
      s1 = wmma_bf16(qfrag[c], b1, s1);
    }

    // ---- causal mask + online softmax ----
#pragma unroll
    for (int r = 0; r < 8; ++r) {
      int qi = qtile + rowoff + r;
      if (kt + ncol > qi)      s0[r] = -3.0e38f;
      if (kt + 16 + ncol > qi) s1[r] = -3.0e38f;
    }
#pragma unroll
    for (int r = 0; r < 8; ++r) {
      float mx = fmaxf(s0[r], s1[r]);
#pragma unroll
      for (int off = 8; off >= 1; off >>= 1) mx = fmaxf(mx, __shfl_xor(mx, off, 32));
      float mnew = fmaxf(mrow[r], mx);
      float corr = __expf(mrow[r] - mnew);
      lrow[r] *= corr;
#pragma unroll
      for (int c = 0; c < 8; ++c) oacc[c][r] *= corr;
      float p0 = __expf(s0[r] - mnew);
      float p1 = __expf(s1[r] - mnew);
      lrow[r] += p0 + p1;
      mrow[r] = mnew;
      P[(rowoff + r) * 32 + ncol]      = (__bf16)p0;
      P[(rowoff + r) * 32 + 16 + ncol] = (__bf16)p1;
    }
    __syncthreads();

    // ---- O += P * V  (contract over 32 keys) ----
    v16bf pfrag = load_a_frag(P, 32, 0, 0, lane);
#pragma unroll
    for (int c = 0; c < 8; ++c) {
      v16bf vb = load_b_frag(Vh, T, c * 16, kt, lane);
      oacc[c] = wmma_bf16(pfrag, vb, oacc[c]);
    }
    __syncthreads();
  }

  // ---- normalize and store ----
#pragma unroll
  for (int r = 0; r < 8; ++r) {
    float l = lrow[r];
#pragma unroll
    for (int off = 8; off >= 1; off >>= 1) l += __shfl_xor(l, off, 32);
    float invl = 1.0f / l;
    int row = qtile + rowoff + r;
    __bf16* orow = aout + (size_t)row * DIM + (size_t)h * HD;
#pragma unroll
    for (int c = 0; c < 8; ++c)
      orow[c * 16 + ncol] = (__bf16)(oacc[c][r] * invl);
  }
}

// ---------------------------------------------------------------------------
// Host orchestration
// ---------------------------------------------------------------------------
extern "C" void kernel_launch(void* const* d_in, const int* in_sizes, int n_in,
                              void* d_out, int out_size, void* d_ws, size_t ws_size,
                              hipStream_t stream) {
  const float* x  = (const float*)d_in[0];
  const float* Wq = (const float*)d_in[1];
  const float* Wk = (const float*)d_in[2];
  const float* Wv = (const float*)d_in[3];
  const float* Wo = (const float*)d_in[4];
  const float* qg = (const float*)d_in[5];
  float* out = (float*)d_out;

  const int T  = in_sizes[0] / DIM;   // 4096
  const int KV = NKV * HD;            // 512

  char* ws = (char*)d_ws;
  size_t off = 0;
  auto carve = [&](size_t bytes) -> void* {
    void* p = ws + off;
    off += (bytes + 255) & ~(size_t)255;
    return p;
  };
  __bf16* xb  = (__bf16*)carve((size_t)T * DIM * 2);
  __bf16* Wqb = (__bf16*)carve((size_t)DIM * DIM * 2);
  __bf16* Wkb = (__bf16*)carve((size_t)KV * DIM * 2);
  __bf16* Wvb = (__bf16*)carve((size_t)KV * DIM * 2);
  __bf16* Wob = (__bf16*)carve((size_t)DIM * DIM * 2);
  float*  qf  = (float*) carve((size_t)T * DIM * 4);
  float*  kf  = (float*) carve((size_t)T * KV * 4);
  float*  vf  = (float*) carve((size_t)T * KV * 4);
  __bf16* Qb  = (__bf16*)carve((size_t)NH  * T * HD * 2);
  __bf16* Kb  = (__bf16*)carve((size_t)NKV * T * HD * 2);
  __bf16* Vt  = (__bf16*)carve((size_t)NKV * HD * T * 2);
  __bf16* aout = (__bf16*)qf;  // qf is dead once Qb is built; reuse as aout

  // 1) convert inputs to bf16
  {
    int n;
    n = T * DIM;   f32_to_bf16_kernel<<<(n + 255) / 256, 256, 0, stream>>>(x,  xb,  n);
    n = DIM * DIM; f32_to_bf16_kernel<<<(n + 255) / 256, 256, 0, stream>>>(Wq, Wqb, n);
    n = KV * DIM;  f32_to_bf16_kernel<<<(n + 255) / 256, 256, 0, stream>>>(Wk, Wkb, n);
    n = KV * DIM;  f32_to_bf16_kernel<<<(n + 255) / 256, 256, 0, stream>>>(Wv, Wvb, n);
    n = DIM * DIM; f32_to_bf16_kernel<<<(n + 255) / 256, 256, 0, stream>>>(Wo, Wob, n);
  }

  // 2) QKV projections (TDM-staged WMMA GEMM)
  gemm_bf16_kernel<<<dim3(DIM / 64, T / 64), 128, 0, stream>>>(xb, Wqb, qf, T, DIM, DIM);
  gemm_bf16_kernel<<<dim3(KV  / 64, T / 64), 128, 0, stream>>>(xb, Wkb, kf, T, KV,  DIM);
  gemm_bf16_kernel<<<dim3(KV  / 64, T / 64), 128, 0, stream>>>(xb, Wvb, vf, T, KV,  DIM);

  // 3) RMSNorm + RoPE (+gain*HEAD_DIM^-1/2 folded into Q); pack bf16 [h][T][128]
  rope_rms_pack_kernel<<<(T * NH)  / 8, 256, 0, stream>>>(qf, qg,      Qb, T, NH,  0.08838834764831845f);
  rope_rms_pack_kernel<<<(T * NKV) / 8, 256, 0, stream>>>(kf, nullptr, Kb, T, NKV, 1.0f);

  // 4) V -> transposed bf16 [h][128][T]
  {
    int n = T * KV;
    v_transpose_kernel<<<(n + 255) / 256, 256, 0, stream>>>(vf, Vt, T);
  }

  // 5) flash attention (WMMA), writes aout bf16 [T, 2048]
  attention_kernel<<<dim3(T / 16, NH), 32, 0, stream>>>(Qb, Kb, Vt, aout, T);

  // 6) output projection (TDM-staged WMMA GEMM) -> fp32 d_out
  gemm_bf16_kernel<<<dim3(DIM / 64, T / 64), 128, 0, stream>>>(aout, Wob, out, T, DIM, DIM);
}
